// ScatteringPConv_54640573940009
// MI455X (gfx1250) — matrix operand order repaired and checked by reading, not compiled, not run on hardware
//
#include <hip/hip_runtime.h>
#include <stdint.h>

typedef __attribute__((ext_vector_type(16))) __bf16 v16bf;
typedef __attribute__((ext_vector_type(8)))  float  v8f;

#define C_IN   32
#define H_DIM  256
#define W_DIM  256
#define F_NUM  16
#define KH     7
#define O_OUT  256
#define K_DIM  512     // C_IN * F_NUM
#define N_DIM  65536   // H * W
#define KT_NUM 16      // K_DIM / 32
#define NT_NUM 4096    // N_DIM / 16
#define EPSV   1e-8f

// round-to-nearest-even f32 -> bf16 (bit trick; avoids relying on __bf16 arith)
__device__ __forceinline__ uint32_t f2bf(float f) {
  uint32_t u = __float_as_uint(f);
  uint32_t r = ((u >> 16) & 1u) + 0x7FFFu;
  return (u + r) >> 16;
}

// -----------------------------------------------------------------------------
// Stage 1: fused masked depthwise 7x7 conv (partial-conv renorm + ReLU),
// writing activations as bf16 directly in WMMA B-fragment layout:
//   B (32x16, bf16): lanes 0-15 <-> N, hold K=0..15 (VGPR r = pair 2r,2r+1);
//                    lanes 16-31 hold K=16..31.
// k = c*16 + f  =>  lane half selected by (c&1), dword r = f/2, half = f&1.
// Fragment storage (dwords): [nt][kt][lane][r]  (lane-contiguous 32B per lane)
// -----------------------------------------------------------------------------
__global__ void __launch_bounds__(256)
stage1_kernel(const float* __restrict__ x, const float* __restrict__ mask,
              const float* __restrict__ fb, uint32_t* __restrict__ a_frag,
              float* __restrict__ scale2) {
  __shared__ float sfilt[F_NUM * KH * KH];  // 784
  __shared__ float sm[22 * 22];
  __shared__ float sxm[22 * 22];

  const int tx  = threadIdx.x;         // pixel col in tile (0..15) == N%16
  const int ty  = threadIdx.y;         // pixel row in tile
  const int tid = ty * 16 + tx;
  const int w0  = blockIdx.x * 16;
  const int h0  = blockIdx.y * 16;

  for (int i = tid; i < F_NUM * KH * KH; i += 256) sfilt[i] = fb[i];

  const int hh = h0 + ty;
  const int ww = w0 + tx;
  const int nt = (hh * W_DIM + ww) >> 4;   // 16-wide N tile index
  int validCount = 0;

  for (int c = 0; c < C_IN; ++c) {
    __syncthreads();  // protect LDS tiles from previous iteration
    // 22x22 halo tile of mask and x*mask, zero-padded ('SAME')
    for (int i = tid; i < 22 * 22; i += 256) {
      int ih = h0 - 3 + i / 22;
      int iw = w0 - 3 + i % 22;
      float m = 0.f, v = 0.f;
      if (ih >= 0 && ih < H_DIM && iw >= 0 && iw < W_DIM) {
        int gi = c * (H_DIM * W_DIM) + ih * W_DIM + iw;
        m = mask[gi];
        v = m * x[gi];
      }
      sm[i]  = m;
      sxm[i] = v;
    }
    __syncthreads();

    float msum = 0.f;
    float raw[F_NUM];
#pragma unroll
    for (int f = 0; f < F_NUM; ++f) raw[f] = 0.f;

    for (int dy = 0; dy < KH; ++dy) {
#pragma unroll
      for (int dx = 0; dx < KH; ++dx) {
        int   si = (ty + dy) * 22 + (tx + dx);
        float m  = sm[si];
        float v  = sxm[si];
        msum += m;
#pragma unroll
        for (int f = 0; f < F_NUM; ++f)
          raw[f] = fmaf(v, sfilt[f * 49 + dy * KH + dx], raw[f]);
      }
    }

    const bool  valid = msum > 0.f;
    validCount += valid ? 1 : 0;
    const float sc = valid ? (49.f / fmaxf(msum, EPSV)) : 0.f;

    uint32_t dws[8];
#pragma unroll
    for (int r = 0; r < 8; ++r) {
      float a0 = fmaxf(raw[2 * r] * sc, 0.f);       // relu(y), zero if invalid
      float a1 = fmaxf(raw[2 * r + 1] * sc, 0.f);
      dws[r] = f2bf(a0) | (f2bf(a1) << 16);
    }
    const int lane = tx + (c & 1) * 16;
    const int base = ((nt * KT_NUM + (c >> 1)) * 32 + lane) * 8;  // dword idx
    uint4* p = (uint4*)(a_frag + base);
    p[0] = make_uint4(dws[0], dws[1], dws[2], dws[3]);
    p[1] = make_uint4(dws[4], dws[5], dws[6], dws[7]);
  }

  // msum2 = F * (#valid channels); z-scale = 512/msum2 (0 where no valid)
  const float msum2 = (float)F_NUM * (float)validCount;
  const float s2 = (validCount > 0) ? ((float)K_DIM / fmaxf(msum2, EPSV)) : 0.f;
  scale2[hh * W_DIM + ww] = s2;
}

// -----------------------------------------------------------------------------
// Transcode w_out (256x512 fp32) into bf16 WMMA A-fragment layout (16-bit A
// 16x32, this code's contiguous k order, same as B):
//   kk<16: lane=m,    r=kk/2        kk>=16   : lane=m+16, r=(kk-16)/2
// Fragment storage (dwords): [mt][kt][lane][r]
// -----------------------------------------------------------------------------
__global__ void __launch_bounds__(256)
wconv_kernel(const float* __restrict__ w, uint16_t* __restrict__ w_frag) {
  const int idx = blockIdx.x * 256 + threadIdx.x;   // 0 .. 256*512-1
  const int m = idx >> 9;
  const int k = idx & 511;
  const int mt = m >> 4, ml = m & 15;
  const int kt = k >> 5, kk = k & 31;
  int lane, r;
  if (kk < 16)       { lane = ml;      r = kk >> 1; }
  else               { lane = ml + 16; r = (kk - 16) >> 1; }
  const int half = kk & 1;
  const int e = (((mt * KT_NUM + kt) * 32 + lane) * 8 + r) * 2 + half;
  w_frag[e] = (uint16_t)f2bf(w[idx]);
}

// -----------------------------------------------------------------------------
// Stage 2: GEMM z[256 x 65536] = W[256x512] * A[512x65536] via
// v_wmma_f32_16x16x32_bf16, + per-pixel partial-conv rescale.
// Block = 8 waves, 1 block per 16-wide N column; each wave owns 2 M tiles and
// shares the B fragment between the two WMMAs of each K step.
// -----------------------------------------------------------------------------
__global__ void __launch_bounds__(256)
gemm_kernel(const __bf16* __restrict__ a_frag, const __bf16* __restrict__ w_frag,
            const float* __restrict__ scale2, float* __restrict__ out) {
  const int nt   = blockIdx.x;
  const int wave = threadIdx.x >> 5;
  const int lane = threadIdx.x & 31;
  const int mt0  = wave * 2;
  const int mt1  = mt0 + 1;

  v8f c0 = {};
  v8f c1 = {};

  const __bf16* bbase  = a_frag + ((size_t)nt  * KT_NUM * 32 + lane) * 16;
  const __bf16* abase0 = w_frag + ((size_t)mt0 * KT_NUM * 32 + lane) * 16;
  const __bf16* abase1 = w_frag + ((size_t)mt1 * KT_NUM * 32 + lane) * 16;

  for (int kt = 0; kt < KT_NUM; ++kt) {
    v16bf b  = *(const v16bf*)(bbase  + (size_t)kt * 512);  // 32 lanes * 16 elems
    v16bf a0 = *(const v16bf*)(abase0 + (size_t)kt * 512);
    v16bf a1 = *(const v16bf*)(abase1 + (size_t)kt * 512);
    if (kt + 1 < KT_NUM)
      __builtin_prefetch(bbase + (size_t)(kt + 1) * 512, 0, 3);  // global_prefetch_b8
    // (neg_a, A, neg_b, B, c_mod, C, reuse_a, reuse_b)
    c0 = __builtin_amdgcn_wmma_f32_16x16x32_bf16(false, a0, false, b,
                                                 (short)0, c0, false, false);
    c1 = __builtin_amdgcn_wmma_f32_16x16x32_bf16(false, a1, false, b,
                                                 (short)0, c1, false, false);
  }

  // C/D layout: VGPR v, lanes 0-15 -> M=v, N=lane; lanes 16-31 -> M=v+8.
  const int   nl = lane & 15;
  const float s  = scale2[nt * 16 + nl];
  const int   mb0 = mt0 * 16 + ((lane >> 4) << 3);
  const int   mb1 = mt1 * 16 + ((lane >> 4) << 3);
#pragma unroll
  for (int v = 0; v < 8; ++v) {
    out[(size_t)(mb0 + v) * N_DIM + nt * 16 + nl] = c0[v] * s;
    out[(size_t)(mb1 + v) * N_DIM + nt * 16 + nl] = c1[v] * s;
  }
}

extern "C" void kernel_launch(void* const* d_in, const int* in_sizes, int n_in,
                              void* d_out, int out_size, void* d_ws, size_t ws_size,
                              hipStream_t stream) {
  const float* x    = (const float*)d_in[0];  // (32,256,256)
  const float* mask = (const float*)d_in[1];  // (32,256,256)
  const float* fb   = (const float*)d_in[2];  // (16,7,7)
  const float* w    = (const float*)d_in[3];  // (256,512,1,1)
  float* out = (float*)d_out;                 // (256,256,256)

  uint8_t* ws = (uint8_t*)d_ws;
  const size_t A_BYTES = (size_t)K_DIM * N_DIM * 2;   // 64 MB bf16 activations
  const size_t W_BYTES = (size_t)O_OUT * K_DIM * 2;   // 256 KB bf16 weights
  uint32_t* a_frag = (uint32_t*)ws;
  uint16_t* w_frag = (uint16_t*)(ws + A_BYTES);
  float*    scale2 = (float*)(ws + A_BYTES + W_BYTES);

  wconv_kernel<<<(O_OUT * K_DIM) / 256, 256, 0, stream>>>(w, w_frag);
  stage1_kernel<<<dim3(W_DIM / 16, H_DIM / 16), dim3(16, 16), 0, stream>>>(
      x, mask, fb, a_frag, scale2);
  gemm_kernel<<<NT_NUM, 256, 0, stream>>>((const __bf16*)a_frag,
                                          (const __bf16*)w_frag, scale2, out);
}
